// Qwen35Experts_17394617548826
// MI455X (gfx1250) — matrix operand order repaired and checked by reading, not compiled, not run on hardware
//
#include <hip/hip_runtime.h>

// ---------------------------------------------------------------------------
// Types for CDNA5 WMMA
// ---------------------------------------------------------------------------
typedef __attribute__((ext_vector_type(16))) __bf16 v16bf;
typedef __attribute__((ext_vector_type(8)))  __bf16 v8bf;
typedef __attribute__((ext_vector_type(8)))  float  v8f;

union Pack8 {
    __bf16 b[8];
    uint4  u;
};

// Problem constants (match reference setup_inputs)
#define T_TOK 4096
#define D_DIM 1024
#define I_DIM 512
#define E_EXP 64
#define K_TOP 8
#define P_PAIRS (T_TOK * K_TOP)
#define C_CAP 1024            // 2 * (P / E)

#define BM 128
#define BN 64
#define BK 32

// ---------------------------------------------------------------------------
// CDNA5 async global->LDS copy (16 bytes per lane), tracked on ASYNCcnt.
// ldsoff: LDS byte address (low 32 bits of generic shared pointer);
// gptr:   64-bit global address (VGPR pair).
// ---------------------------------------------------------------------------
__device__ inline void async_copy_b128(unsigned ldsoff, const void* gptr) {
    asm volatile("global_load_async_to_lds_b128 %0, %1, off"
                 :: "v"(ldsoff), "v"(gptr)
                 : "memory");
}

__device__ inline void wait_async0() {
    asm volatile("s_wait_asynccnt 0x0" ::: "memory");
}

// ---------------------------------------------------------------------------
// Fragment loaders (LDS, row-major bf16 tiles, row stride = BK)
// A 16x32 subtile: lane = row m (lane&15); K halves split by lane>>4
// ---------------------------------------------------------------------------
__device__ inline v16bf load_a_frag(const __bf16* at, int lane) {
    int m = lane & 15;
    int h = (lane >> 4) * 16;            // 0: K 0-15; 1: K 16-31
    return *(const v16bf*)(at + m * BK + h);
}

// B 32x16 subtile stored as 16 n-rows x 32 k: lane = col n (lane&15),
// K half selected by lane>>4 -> one contiguous 32-byte read.
__device__ inline v16bf load_b_frag(const __bf16* bt, int lane) {
    int n  = lane & 15;
    int kh = (lane >> 4) * 16;
    return *(const v16bf*)(bt + n * BK + kh);
}

// ---------------------------------------------------------------------------
// Kernel 0: zero the output (we accumulate into it)
// ---------------------------------------------------------------------------
__global__ void k_zero(float* out, int n) {
    int i = blockIdx.x * blockDim.x + threadIdx.x;
    if (i < n) out[i] = 0.0f;
}

// ---------------------------------------------------------------------------
// Kernel 1: deterministic routing. One wave (32 lanes) per expert scans all
// pairs in order; ballot+prefix keeps the reference's stable ordering.
// ---------------------------------------------------------------------------
__global__ void k_route(const int* __restrict__ topk_idx,
                        const float* __restrict__ topk_w,
                        int* __restrict__ cnt,
                        int* __restrict__ tok,
                        float* __restrict__ wv) {
    int e = blockIdx.x;
    int lane = threadIdx.x;
    int slot = 0;
    for (int base = 0; base < P_PAIRS; base += 32) {
        int p = base + lane;
        bool m = (p < P_PAIRS) && (topk_idx[p] == e);
        unsigned mask = (unsigned)__ballot(m);
        int prefix = __popc(mask & ((1u << lane) - 1u));
        if (m) {
            int s = slot + prefix;
            if (s < C_CAP) {
                tok[e * C_CAP + s] = p / K_TOP;
                wv[e * C_CAP + s]  = topk_w[p];
            }
        }
        slot += __popc(mask);
    }
    if (lane == 0) cnt[e] = (slot < C_CAP) ? slot : C_CAP;
}

// ---------------------------------------------------------------------------
// Kernel 2: gather tokens into per-expert bf16 buffers, zero-fill empty slots
// ---------------------------------------------------------------------------
__global__ void k_gather(const float* __restrict__ hs,
                         const int* __restrict__ cnt,
                         const int* __restrict__ tok,
                         __bf16* __restrict__ Xb) {
    int row = blockIdx.x;                 // e*C + s
    int e = row / C_CAP, s = row % C_CAP;
    size_t base = (size_t)row * D_DIM;
    if (s < cnt[e]) {
        const float* src = hs + (size_t)tok[row] * D_DIM;
        for (int i = threadIdx.x; i < D_DIM; i += blockDim.x)
            Xb[base + i] = (__bf16)src[i];
    } else {
        for (int i = threadIdx.x; i < D_DIM; i += blockDim.x)
            Xb[base + i] = (__bf16)0.0f;
    }
}

// ---------------------------------------------------------------------------
// Kernel 3: grouped GEMM1 + fused SiLU-gate.
// h[c, i] = silu(x.Wg^T) * (x.Wu^T), per expert. gate rows [0,I), up rows [I,2I).
// A tile staged with CDNA5 async global->LDS; B tiles converted fp32->bf16.
// ---------------------------------------------------------------------------
__global__ __launch_bounds__(256)
void k_gemm1(const __bf16* __restrict__ Xb,
             const float* __restrict__ gup,
             const int* __restrict__ cnt,
             __bf16* __restrict__ Hb) {
    int e  = blockIdx.z;
    int m0 = blockIdx.x * BM;
    int n0 = blockIdx.y * BN;
    if (m0 >= cnt[e]) return;             // uniform: EXEC stays all-1 for WMMA

    __shared__ __align__(16) __bf16 As[BM * BK];
    __shared__ __align__(16) __bf16 Bg[BN * BK];
    __shared__ __align__(16) __bf16 Bu[BN * BK];

    int tid = threadIdx.x, lane = tid & 31, wave = tid >> 5;
    int wm = (wave & 3) * 32;             // 4 waves along M
    int wn = (wave >> 2) * 32;            // 2 waves along N

    v8f accg[2][2] = {};
    v8f accu[2][2] = {};

    const __bf16* Abase = Xb + ((size_t)e * C_CAP + m0) * D_DIM;
    const float*  Gbase = gup + (size_t)e * (2 * I_DIM) * D_DIM;

    // per-thread A staging coordinates (two 16B chunks per thread)
    int ar0 = (tid * 2) >> 2, ac0 = ((tid * 2) & 3) * 8;
    int ar1 = (tid * 2 + 1) >> 2, ac1 = ((tid * 2 + 1) & 3) * 8;
    unsigned aoff0 = (unsigned)(uintptr_t)&As[ar0 * BK + ac0];
    unsigned aoff1 = (unsigned)(uintptr_t)&As[ar1 * BK + ac1];

    // per-thread B staging coordinates (64 rows x 32 cols, 8 elems/thread)
    int br = tid >> 2;
    int bc = (tid & 3) * 8;
    const float* gsrc0 = Gbase + (size_t)(n0 + br) * D_DIM + bc;
    const float* usrc0 = Gbase + (size_t)(I_DIM + n0 + br) * D_DIM + bc;

    for (int k0 = 0; k0 < D_DIM; k0 += BK) {
        // stage A: async global->LDS (bf16 passthrough, no VGPR round-trip)
        async_copy_b128(aoff0, Abase + (size_t)ar0 * D_DIM + k0 + ac0);
        async_copy_b128(aoff1, Abase + (size_t)ar1 * D_DIM + k0 + ac1);

        // stage B gate/up with on-the-fly fp32->bf16 conversion
        {
            const float* gsrc = gsrc0 + k0;
            const float* usrc = usrc0 + k0;
            if (k0 + BK < D_DIM) {
                __builtin_prefetch(gsrc + BK, 0, 3);   // global_prefetch_b8 (near)
                __builtin_prefetch(usrc + BK, 0, 3);
            }
            Pack8 pg, pu;
#pragma unroll
            for (int i = 0; i < 8; ++i) {
                pg.b[i] = (__bf16)gsrc[i];
                pu.b[i] = (__bf16)usrc[i];
            }
            *(uint4*)&Bg[br * BK + bc] = pg.u;
            *(uint4*)&Bu[br * BK + bc] = pu.u;
        }
        wait_async0();                    // A tile landed in LDS
        __syncthreads();

        v16bf a[2], bg[2], bu[2];
#pragma unroll
        for (int i = 0; i < 2; ++i) a[i]  = load_a_frag(&As[(wm + i * 16) * BK], lane);
#pragma unroll
        for (int j = 0; j < 2; ++j) {
            bg[j] = load_b_frag(&Bg[(wn + j * 16) * BK], lane);
            bu[j] = load_b_frag(&Bu[(wn + j * 16) * BK], lane);
        }
#pragma unroll
        for (int i = 0; i < 2; ++i)
#pragma unroll
            for (int j = 0; j < 2; ++j) {
                accg[i][j] = __builtin_amdgcn_wmma_f32_16x16x32_bf16(
                    false, a[i], false, bg[j], (short)0, accg[i][j], false, false);
                accu[i][j] = __builtin_amdgcn_wmma_f32_16x16x32_bf16(
                    false, a[i], false, bu[j], (short)0, accu[i][j], false, false);
            }
        __syncthreads();
    }

    // fused activation + store h as bf16 (fast sigmoid: exp + v_rcp_f32)
    int mlane = (lane >> 4) * 8;
    int nlane = lane & 15;
#pragma unroll
    for (int i = 0; i < 2; ++i)
#pragma unroll
        for (int j = 0; j < 2; ++j)
#pragma unroll
            for (int r = 0; r < 8; ++r) {
                float g = accg[i][j][r];
                float u = accu[i][j][r];
                float sig = __builtin_amdgcn_rcpf(1.0f + __expf(-g));
                float h = g * sig * u;                 // silu(g)*u
                int mrow = m0 + wm + i * 16 + mlane + r;
                int ncol = n0 + wn + j * 16 + nlane;
                Hb[((size_t)e * C_CAP + mrow) * I_DIM + ncol] = (__bf16)h;
            }
}

// ---------------------------------------------------------------------------
// Kernel 4: grouped GEMM2 + fused weighted scatter-add to token order.
// y[c, d] = sum_i h[c,i] * down[e, d, i];  out[tok] += w * y
// ---------------------------------------------------------------------------
__global__ __launch_bounds__(256)
void k_gemm2(const __bf16* __restrict__ Hb,
             const float* __restrict__ dwn,
             const int* __restrict__ cnt,
             const int* __restrict__ tok,
             const float* __restrict__ wv,
             float* __restrict__ out) {
    int e  = blockIdx.z;
    int m0 = blockIdx.x * BM;
    int n0 = blockIdx.y * BN;
    int ce = cnt[e];
    if (m0 >= ce) return;

    __shared__ __align__(16) __bf16 As[BM * BK];
    __shared__ __align__(16) __bf16 Bs[BN * BK];

    int tid = threadIdx.x, lane = tid & 31, wave = tid >> 5;
    int wm = (wave & 3) * 32;
    int wn = (wave >> 2) * 32;

    v8f acc[2][2] = {};

    const __bf16* Abase = Hb + ((size_t)e * C_CAP + m0) * I_DIM;
    const float*  Dbase = dwn + (size_t)e * D_DIM * I_DIM;

    int ar0 = (tid * 2) >> 2, ac0 = ((tid * 2) & 3) * 8;
    int ar1 = (tid * 2 + 1) >> 2, ac1 = ((tid * 2 + 1) & 3) * 8;
    unsigned aoff0 = (unsigned)(uintptr_t)&As[ar0 * BK + ac0];
    unsigned aoff1 = (unsigned)(uintptr_t)&As[ar1 * BK + ac1];

    int br = tid >> 2;
    int bc = (tid & 3) * 8;
    const float* dsrc0 = Dbase + (size_t)(n0 + br) * I_DIM + bc;

    for (int k0 = 0; k0 < I_DIM; k0 += BK) {
        // stage A: async global->LDS (bf16 h buffer)
        async_copy_b128(aoff0, Abase + (size_t)ar0 * I_DIM + k0 + ac0);
        async_copy_b128(aoff1, Abase + (size_t)ar1 * I_DIM + k0 + ac1);

        {
            const float* dsrc = dsrc0 + k0;
            if (k0 + BK < I_DIM) __builtin_prefetch(dsrc + BK, 0, 3);
            Pack8 pd;
#pragma unroll
            for (int i = 0; i < 8; ++i) pd.b[i] = (__bf16)dsrc[i];
            *(uint4*)&Bs[br * BK + bc] = pd.u;
        }
        wait_async0();
        __syncthreads();

        v16bf a[2], b[2];
#pragma unroll
        for (int i = 0; i < 2; ++i) a[i] = load_a_frag(&As[(wm + i * 16) * BK], lane);
#pragma unroll
        for (int j = 0; j < 2; ++j) b[j] = load_b_frag(&Bs[(wn + j * 16) * BK], lane);
#pragma unroll
        for (int i = 0; i < 2; ++i)
#pragma unroll
            for (int j = 0; j < 2; ++j)
                acc[i][j] = __builtin_amdgcn_wmma_f32_16x16x32_bf16(
                    false, a[i], false, b[j], (short)0, acc[i][j], false, false);
        __syncthreads();
    }

    // weighted scatter-add back to token order (guarded by expert count)
    int mlane = (lane >> 4) * 8;
    int nlane = lane & 15;
#pragma unroll
    for (int i = 0; i < 2; ++i)
#pragma unroll
        for (int j = 0; j < 2; ++j)
#pragma unroll
            for (int r = 0; r < 8; ++r) {
                int s = m0 + wm + i * 16 + mlane + r;
                if (s < ce) {
                    int   t = tok[e * C_CAP + s];
                    float w = wv[e * C_CAP + s];
                    int   d = n0 + wn + j * 16 + nlane;
                    atomicAdd(&out[(size_t)t * D_DIM + d], w * acc[i][j][r]);
                }
            }
}

// ---------------------------------------------------------------------------
// Launch
// ---------------------------------------------------------------------------
extern "C" void kernel_launch(void* const* d_in, const int* in_sizes, int n_in,
                              void* d_out, int out_size, void* d_ws, size_t ws_size,
                              hipStream_t stream) {
    const float* hidden = (const float*)d_in[0];
    const int*   tidx   = (const int*)d_in[1];
    const float* tw     = (const float*)d_in[2];
    const float* gup    = (const float*)d_in[3];
    const float* dwn    = (const float*)d_in[4];
    float* out = (float*)d_out;

    // workspace layout
    char* w = (char*)d_ws;
    int*   cnt = (int*)w;             w += 256;
    int*   tok = (int*)w;             w += (size_t)E_EXP * C_CAP * sizeof(int);
    float* wv  = (float*)w;           w += (size_t)E_EXP * C_CAP * sizeof(float);
    __bf16* Xb = (__bf16*)w;          w += (size_t)E_EXP * C_CAP * D_DIM * sizeof(__bf16);
    __bf16* Hb = (__bf16*)w;          w += (size_t)E_EXP * C_CAP * I_DIM * sizeof(__bf16);

    int n_out = T_TOK * D_DIM;
    k_zero<<<(n_out + 255) / 256, 256, 0, stream>>>(out, n_out);

    k_route<<<E_EXP, 32, 0, stream>>>(tidx, tw, cnt, tok, wv);

    k_gather<<<E_EXP * C_CAP, 256, 0, stream>>>(hidden, cnt, tok, Xb);

    dim3 g1(C_CAP / BM, I_DIM / BN, E_EXP);   // (8, 8, 64)
    k_gemm1<<<g1, 256, 0, stream>>>(Xb, gup, cnt, Hb);

    dim3 g2(C_CAP / BM, D_DIM / BN, E_EXP);   // (8, 16, 64)
    k_gemm2<<<g2, 256, 0, stream>>>(Hb, dwn, cnt, tok, wv, out);
}
